// SS2D_1_32238024524048
// MI455X (gfx1250) — matrix operand order repaired and checked by reading, not compiled, not run on hardware
//
#include <hip/hip_runtime.h>

typedef __attribute__((ext_vector_type(16))) _Float16 v16h;
typedef __attribute__((ext_vector_type(8)))  _Float16 v8h;
typedef __attribute__((ext_vector_type(8)))  float    v8f;
typedef __attribute__((ext_vector_type(4)))  float    v4f;

#define LPIX 4096
#define DIN  192

__device__ __forceinline__ float siluf(float v) { return v / (1.f + __expf(-v)); }
__device__ __forceinline__ float softplusf(float x) { return (x > 30.f) ? x : log1pf(__expf(x)); }

// ---------------------------------------------------------------------------
// Pack A: f32 [M x K] row-major -> f16 [Mp x Kp] row-major, zero padded.
// ---------------------------------------------------------------------------
__global__ void pack_a_kernel(const float* __restrict__ src, _Float16* __restrict__ dst,
                              int M, int K, int Mp, int Kp, long sS, long sD)
{
    int idx = blockIdx.x * blockDim.x + threadIdx.x;
    if (idx >= Mp * Kp) return;
    int m = idx / Kp, k = idx - m * Kp;
    const float* s = src + (long)blockIdx.y * sS;
    _Float16* d = dst + (long)blockIdx.y * sD;
    d[idx] = (m < M && k < K) ? (_Float16)s[(long)m * K + k] : (_Float16)0.f;
}

// ---------------------------------------------------------------------------
// Pack B: f32 [K x N] row-major -> f16 K-panels Bp[kt][n][kk] (kk contiguous),
// K padded to Kp (multiple of 32), zero filled.
// ---------------------------------------------------------------------------
__global__ void pack_b_kernel(const float* __restrict__ src, _Float16* __restrict__ dst,
                              int K, int N, int Kp, long sS, long sD)
{
    int idx = blockIdx.x * blockDim.x + threadIdx.x;
    if (idx >= Kp * N) return;
    int kk = idx & 31;
    int rest = idx >> 5;                 // kt*N + n
    int n = rest % N, kt = rest / N;
    int k = kt * 32 + kk;
    const float* s = src + (long)blockIdx.y * sS;
    _Float16* d = dst + (long)blockIdx.y * sD;
    d[idx] = (k < K) ? (_Float16)s[(long)k * N + n] : (_Float16)0.f;
}

// ---------------------------------------------------------------------------
// WMMA GEMM, pre-packed f16 operands, f32 accumulate.
// A: f16 [Mp x KP] row-major.  B: f16 panels [KP/32][N][32].  C: f32 [Mp x N].
// One wave computes a 16x32 output tile (A fragment reused across 2 WMMAs).
// blockDim=128 (4 waves); grid=(N/128, Mp/16, batch). No guards; K unrolled.
// ---------------------------------------------------------------------------
template <int KP, bool BIAS>
__global__ void wmma_gemm_f16_kernel(const _Float16* __restrict__ A,
                                     const _Float16* __restrict__ B,
                                     float* __restrict__ C, const float* __restrict__ bias,
                                     int N, long strideA, long strideB, long strideC)
{
    const int lane  = threadIdx.x & 31;
    const int waveB = threadIdx.x >> 5;
    const int tile2 = blockIdx.x * 4 + waveB;     // 32-column tile index
    const int tileM = blockIdx.y;
    const int kb    = blockIdx.z;
    A += (long)kb * strideA;
    B += (long)kb * strideB;
    C += (long)kb * strideC;

    const int am  = tileM * 16 + (lane & 15);
    const int akb = (lane < 16) ? 0 : 16;
    const int bn0 = tile2 * 32 + (lane & 15);
    const int bkb = (lane < 16) ? 0 : 16;

    const _Float16* arow = A + (long)am * KP + akb;

    v8f acc0 = {}, acc1 = {};
#pragma unroll
    for (int k0 = 0; k0 < KP; k0 += 32) {
        v8h a0 = *(const v8h*)(arow + k0);
        v8h a1 = *(const v8h*)(arow + k0 + 8);
        const _Float16* bp0 = B + (((long)(k0 >> 5) * N + bn0) << 5) + bkb;
        v8h b0 = *(const v8h*)(bp0);
        v8h b1 = *(const v8h*)(bp0 + 8);
        v8h b2 = *(const v8h*)(bp0 + (16 << 5));      // +16 columns
        v8h b3 = *(const v8h*)(bp0 + (16 << 5) + 8);
        v16h af  = __builtin_shufflevector(a0, a1, 0,1,2,3,4,5,6,7,8,9,10,11,12,13,14,15);
        v16h bf0 = __builtin_shufflevector(b0, b1, 0,1,2,3,4,5,6,7,8,9,10,11,12,13,14,15);
        v16h bf1 = __builtin_shufflevector(b2, b3, 0,1,2,3,4,5,6,7,8,9,10,11,12,13,14,15);
        acc0 = __builtin_amdgcn_wmma_f32_16x16x32_f16(false, af, false, bf0,
                                                      (short)0, acc0, false, false);
        acc1 = __builtin_amdgcn_wmma_f32_16x16x32_f16(false, af, false, bf1,
                                                      (short)0, acc1, false, false);
    }

    const int mo = (lane < 16) ? 0 : 8;
    v4f bv0 = {}, bv1 = {};
    if (BIAS) {
        const float* bb = bias + tileM * 16 + mo;     // 8 contiguous floats, 16B aligned
        bv0 = *(const v4f*)(bb);
        bv1 = *(const v4f*)(bb + 4);
    }
#pragma unroll
    for (int r = 0; r < 8; ++r) {
        int m = tileM * 16 + mo + r;
        float b = BIAS ? ((r < 4) ? bv0[r & 3] : bv1[r & 3]) : 0.f;
        C[(long)m * N + bn0]      = acc0[r] + b;
        C[(long)m * N + bn0 + 16] = acc1[r] + b;
    }
}

// ---------------------------------------------------------------------------
// Depthwise 3x3 conv + bias + SiLU (pad 1). xz rows 0..191 are the conv input.
// ---------------------------------------------------------------------------
__global__ void dwconv_silu_kernel(const float* __restrict__ xz, const float* __restrict__ cw,
                                   const float* __restrict__ cb, float* __restrict__ xg)
{
    int idx = blockIdx.x * blockDim.x + threadIdx.x;
    if (idx >= DIN * LPIX) return;
    int d = idx >> 12, l = idx & 4095, h = l >> 6, w = l & 63;
    const float* src = xz + (long)d * LPIX;
    const float* k = cw + d * 9;
    float s = cb[d];
#pragma unroll
    for (int i = -1; i <= 1; ++i) {
        int hh = h + i; if (hh < 0 || hh > 63) continue;
#pragma unroll
        for (int j = -1; j <= 1; ++j) {
            int ww = w + j; if (ww < 0 || ww > 63) continue;
            s += src[hh * 64 + ww] * k[(i + 1) * 3 + (j + 1)];
        }
    }
    xg[idx] = siluf(s);
}

// ---------------------------------------------------------------------------
// Build 4-direction xs: f32 rows (for the scan's u input) AND packed f16
// B-panels (for the x_proj WMMA GEMM).
// ---------------------------------------------------------------------------
__global__ void build_xs_kernel(const float* __restrict__ xg, float* __restrict__ xs,
                                _Float16* __restrict__ xsp)
{
    int idx = blockIdx.x * blockDim.x + threadIdx.x;
    if (idx >= DIN * LPIX) return;
    int d = idx >> 12, l = idx & 4095;
    int t  = ((l & 63) << 6) | (l >> 6);
    int fl = 4095 - l;
    int tf = ((fl & 63) << 6) | (fl >> 6);
    const float* row = xg + (long)d * LPIX;
    float v0 = row[l], v1 = row[t], v2 = row[fl], v3 = row[tf];

    const long DL = (long)DIN * LPIX;
    long base = (long)d * LPIX + l;
    xs[base]          = v0;
    xs[DL + base]     = v1;
    xs[2 * DL + base] = v2;
    xs[3 * DL + base] = v3;

    const long PD = 6L * LPIX * 32;      // per-direction panel size (192/32=6 ktiles)
    long pbase = (((long)(d >> 5) * LPIX + l) << 5) + (d & 31);
    xsp[pbase]          = (_Float16)v0;
    xsp[PD + pbase]     = (_Float16)v1;
    xsp[2 * PD + pbase] = (_Float16)v2;
    xsp[3 * PD + pbase] = (_Float16)v3;
}

// ---------------------------------------------------------------------------
// Selective scan, main path. 16-lane half-wave per (k,d); lane = state.
// x_dbl is stored with 48-row padded stride per direction.
// ---------------------------------------------------------------------------
__global__ void scan_kernel(const float* __restrict__ xs, const float* __restrict__ dts,
                            const float* __restrict__ xdbl, const float* __restrict__ A_logs,
                            const float* __restrict__ Ds, const float* __restrict__ dt_bias,
                            float* __restrict__ out_y)
{
    int lane = threadIdx.x & 31;
    int wave = blockIdx.x * (blockDim.x >> 5) + (threadIdx.x >> 5);
    int p = wave * 2 + (lane >> 4);
    if (p >= 768) return;
    int n = lane & 15;
    int k = p / DIN;

    float An  = -__expf(A_logs[p * 16 + n]);
    float Dv  = Ds[p];
    float dtb = dt_bias[p];
    const float* u_row  = xs  + (long)p * LPIX;
    const float* dt_row = dts + (long)p * LPIX;
    const float* B_row  = xdbl + (long)k * 48 * LPIX + (long)(6 + n) * LPIX;
    const float* C_row  = xdbl + (long)k * 48 * LPIX + (long)(22 + n) * LPIX;
    float* y_row = out_y + (long)p * LPIX;

    float h = 0.f;
#pragma unroll 4
    for (int l = 0; l < LPIX; ++l) {
        float dt = softplusf(dt_row[l] + dtb);
        float u  = u_row[l];
        h = __expf(dt * An) * h + (dt * u) * B_row[l];
        float c = C_row[l] * h;
        c += __shfl_xor(c, 1, 32);
        c += __shfl_xor(c, 2, 32);
        c += __shfl_xor(c, 4, 32);
        c += __shfl_xor(c, 8, 32);
        if (n == 0) y_row[l] = c + u * Dv;
    }
}

// ---------------------------------------------------------------------------
// Combine 4 directions back into row-major layout.
// ---------------------------------------------------------------------------
__global__ void combine_kernel(const float* __restrict__ oy, float* __restrict__ ycomb)
{
    int idx = blockIdx.x * blockDim.x + threadIdx.x;
    if (idx >= DIN * LPIX) return;
    int d = idx >> 12, l = idx & 4095;
    int t = ((l & 63) << 6) | (l >> 6);
    const long DL = (long)DIN * LPIX;
    long rb = (long)d * LPIX;
    const float* y0 = oy + rb;
    const float* y1 = oy + DL + rb;
    const float* y2 = oy + 2 * DL + rb;
    const float* y3 = oy + 3 * DL + rb;
    ycomb[idx] = y0[l] + y2[4095 - l] + y1[t] + y3[4095 - t];
}

// ---------------------------------------------------------------------------
// Channel means: one wave per channel d over 4096 pixels.
// ---------------------------------------------------------------------------
__global__ void chan_mean_kernel(const float* __restrict__ xg, float* __restrict__ cmean)
{
    int lane = threadIdx.x & 31;
    int d = blockIdx.x * (blockDim.x >> 5) + (threadIdx.x >> 5);
    if (d >= DIN) return;
    const float* row = xg + (long)d * LPIX;
    float s = 0.f;
    for (int l = lane; l < LPIX; l += 32) s += row[l];
#pragma unroll
    for (int m = 1; m < 32; m <<= 1) s += __shfl_xor(s, m, 32);
    if (lane == 0) cmean[d] = s * (1.f / (float)LPIX);
}

// ---------------------------------------------------------------------------
// Channel branch prep (single block): xcs, xc_dbl, dtc.
// ---------------------------------------------------------------------------
__global__ void chan_prep_kernel(const float* __restrict__ cmean,
                                 const float* __restrict__ cin_w, const float* __restrict__ cin_b,
                                 const float* __restrict__ xcw, const float* __restrict__ dtcw,
                                 float* __restrict__ xcs, float* __restrict__ xcdbl,
                                 float* __restrict__ dtc)
{
    const int tid = threadIdx.x;
    for (int idx = tid; idx < 2 * 4 * 192; idx += blockDim.x) {
        int k = idx / 768, rem = idx - k * 768;
        int ci = rem / 192, j = rem - ci * 192;
        int jj = k ? (191 - j) : j;
        xcs[idx] = cmean[jj] * cin_w[ci] + cin_b[ci];
    }
    __threadfence();
    __syncthreads();
    for (int idx = tid; idx < 2 * 38 * 192; idx += blockDim.x) {
        int k = idx / 7296, rem = idx - k * 7296;
        int r = rem / 192, j = rem - r * 192;
        float s = 0.f;
#pragma unroll
        for (int ci = 0; ci < 4; ++ci)
            s += xcw[k * 152 + r * 4 + ci] * xcs[(k * 4 + ci) * 192 + j];
        xcdbl[idx] = s;
    }
    __threadfence();
    __syncthreads();
    for (int idx = tid; idx < 2 * 4 * 192; idx += blockDim.x) {
        int k = idx / 768, rem = idx - k * 768;
        int ci = rem / 192, j = rem - ci * 192;
        float s = 0.f;
#pragma unroll
        for (int r = 0; r < 6; ++r)
            s += dtcw[k * 24 + ci * 6 + r] * xcdbl[k * 7296 + r * 192 + j];
        dtc[idx] = s;
    }
}

// ---------------------------------------------------------------------------
// Channel-branch selective scan: 8 pairs, 192 steps. One block of 128.
// ---------------------------------------------------------------------------
__global__ void chan_scan_kernel(const float* __restrict__ xcs, const float* __restrict__ xcdbl,
                                 const float* __restrict__ dtc, const float* __restrict__ Ac_logs,
                                 const float* __restrict__ Dsc, const float* __restrict__ dtcb,
                                 float* __restrict__ yc)
{
    int lane = threadIdx.x & 31;
    int wave = threadIdx.x >> 5;
    int p = wave * 2 + (lane >> 4);   // 0..7
    int n = lane & 15;
    int k = p >> 2;

    float An  = -__expf(Ac_logs[p * 16 + n]);
    float Dv  = Dsc[p];
    float dtb = dtcb[p];
    const float* u_row  = xcs + p * 192;
    const float* dt_row = dtc + p * 192;
    const float* B_row  = xcdbl + k * 7296 + (6 + n) * 192;
    const float* C_row  = xcdbl + k * 7296 + (22 + n) * 192;

    float h = 0.f;
#pragma unroll 4
    for (int j = 0; j < 192; ++j) {
        float dt = softplusf(dt_row[j] + dtb);
        float u  = u_row[j];
        h = __expf(dt * An) * h + (dt * u) * B_row[j];
        float c = C_row[j] * h;
        c += __shfl_xor(c, 1, 32);
        c += __shfl_xor(c, 2, 32);
        c += __shfl_xor(c, 4, 32);
        c += __shfl_xor(c, 8, 32);
        if (n == 0) yc[p * 192 + j] = c + u * Dv;
    }
}

// ---------------------------------------------------------------------------
// gate = LN_192( sum_ci (yc[0]+flip(yc[1])) * cout_w + cout_b ). One block.
// ---------------------------------------------------------------------------
__global__ void gate_kernel(const float* __restrict__ yc, const float* __restrict__ cout_w,
                            const float* __restrict__ cout_b, const float* __restrict__ cnw,
                            const float* __restrict__ cnb, float* __restrict__ gate)
{
    __shared__ float sg[192];
    __shared__ float smu, sinv;
    int tid = threadIdx.x;
    if (tid < 192) {
        float g = cout_b[0];
#pragma unroll
        for (int ci = 0; ci < 4; ++ci)
            g += (yc[ci * 192 + tid] + yc[(4 + ci) * 192 + (191 - tid)]) * cout_w[ci];
        sg[tid] = g;
    }
    __syncthreads();
    if (tid == 0) {
        float s = 0.f, ss = 0.f;
        for (int j = 0; j < 192; ++j) { s += sg[j]; ss += sg[j] * sg[j]; }
        float mu = s * (1.f / 192.f);
        float var = ss * (1.f / 192.f) - mu * mu;
        smu = mu; sinv = rsqrtf(var + 1e-5f);
    }
    __syncthreads();
    if (tid < 192) gate[tid] = (sg[tid] - smu) * sinv * cnw[tid] + cnb[tid];
}

// ---------------------------------------------------------------------------
// Per-pixel LayerNorm over 192 channels, * gate[d] * silu(z[d,l]).
// Writes the result directly as packed f16 B-panels for the out_conv GEMM.
// One wave per pixel (6 channels per lane).
// ---------------------------------------------------------------------------
__global__ void ln_mult_kernel(const float* __restrict__ ycomb, const float* __restrict__ onw,
                               const float* __restrict__ onb, const float* __restrict__ gate,
                               const float* __restrict__ xz, _Float16* __restrict__ yfp)
{
    int lane = threadIdx.x & 31;
    int l = blockIdx.x * (blockDim.x >> 5) + (threadIdx.x >> 5);
    if (l >= LPIX) return;
    float v[6];
    float s = 0.f, ss = 0.f;
#pragma unroll
    for (int i = 0; i < 6; ++i) {
        int d = lane + 32 * i;
        float x = ycomb[(long)d * LPIX + l];
        v[i] = x; s += x; ss += x * x;
    }
#pragma unroll
    for (int m = 1; m < 32; m <<= 1) { s += __shfl_xor(s, m, 32); ss += __shfl_xor(ss, m, 32); }
    float mu  = s * (1.f / 192.f);
    float var = ss * (1.f / 192.f) - mu * mu;
    float inv = rsqrtf(var + 1e-5f);
#pragma unroll
    for (int i = 0; i < 6; ++i) {
        int d = lane + 32 * i;
        float ln = (v[i] - mu) * inv * onw[d] + onb[d];
        float z  = xz[(long)(DIN + d) * LPIX + l];
        long pbase = (((long)(d >> 5) * LPIX + l) << 5) + (d & 31);
        yfp[pbase] = (_Float16)(ln * gate[d] * siluf(z));
    }
}

// ---------------------------------------------------------------------------
extern "C" void kernel_launch(void* const* d_in, const int* in_sizes, int n_in,
                              void* d_out, int out_size, void* d_ws, size_t ws_size,
                              hipStream_t stream)
{
    const float* x       = (const float*)d_in[0];
    const float* in_w    = (const float*)d_in[1];
    const float* in_b    = (const float*)d_in[2];
    const float* conv_w  = (const float*)d_in[3];
    const float* conv_b  = (const float*)d_in[4];
    const float* xproj_w = (const float*)d_in[5];
    const float* dt_w    = (const float*)d_in[6];
    const float* dt_b    = (const float*)d_in[7];
    const float* A_logs  = (const float*)d_in[8];
    const float* Ds      = (const float*)d_in[9];
    const float* onw     = (const float*)d_in[10];
    const float* onb     = (const float*)d_in[11];
    const float* cin_w   = (const float*)d_in[12];
    const float* cin_b   = (const float*)d_in[13];
    const float* cout_w  = (const float*)d_in[14];
    const float* cout_b  = (const float*)d_in[15];
    const float* xcw     = (const float*)d_in[16];
    const float* dtcw    = (const float*)d_in[17];
    const float* dtcb    = (const float*)d_in[18];
    const float* Ac_logs = (const float*)d_in[19];
    const float* Dsc     = (const float*)d_in[20];
    const float* cnw     = (const float*)d_in[21];
    const float* cnb     = (const float*)d_in[22];
    const float* out_w   = (const float*)d_in[23];
    const float* out_b   = (const float*)d_in[24];
    float* out = (float*)d_out;
    float* ws  = (float*)d_ws;

    const long DL = (long)DIN * LPIX;            // 786432
    // ---- f32 region ----
    float* xz    = ws;                            // 2*DL
    float* xg    = ws + 2 * DL;                   // DL
    float* xs    = ws + 3 * DL;                   // 4*DL
    float* xdbl  = ws + 7 * DL;                   // 4*48*4096 = DL (padded to 48 rows)
    float* dts   = xdbl + DL;                     // 4*DL
    float* oy    = dts + 4 * DL;                  // 4*DL
    float* ycomb = oy + 4 * DL;                   // DL
    float* cmean = ycomb + DL;                    // 192
    float* xcs   = cmean + 192;                   // 1536
    float* xcdbl = xcs + 1536;                    // 14592
    float* dtc   = xcdbl + 14592;                 // 1536
    float* yc    = dtc + 1536;                    // 1536
    float* gate  = yc + 1536;                     // 192 (+32 slack)
    // ---- f16 region (16-byte aligned: f32 counts above are multiples of 4) ----
    _Float16* h16   = (_Float16*)(gate + 224);
    _Float16* Ap_in = h16;                        // 384*96   = 36864
    _Float16* Ap_xp = Ap_in + 36864;              // 4*48*192 = 36864
    _Float16* Ap_dt = Ap_xp + 36864;              // 4*192*32 = 24576
    _Float16* Ap_ow = Ap_dt + 24576;              // 96*192   = 18432
    _Float16* Bp_x  = Ap_ow + 18432;              // 3*4096*32  = 393216
    _Float16* Bp_xs = Bp_x + 393216;              // 4*6*4096*32 = 3145728
    _Float16* Bp_dt = Bp_xs + 3145728;            // 4*4096*32  = 524288
    _Float16* Bp_yf = Bp_dt + 524288;             // 6*4096*32  = 786432

    const dim3 gblk(128, 1, 1);
    const dim3 eblk(256, 1, 1);
    const int egrid = (DIN * LPIX) / 256;         // 3072

    // ---- pack weight (A) operands ----
    pack_a_kernel<<<dim3(144, 1), eblk, 0, stream>>>(in_w, Ap_in, 384, 96, 384, 96, 0, 0);
    pack_a_kernel<<<dim3(36, 4),  eblk, 0, stream>>>(xproj_w, Ap_xp, 38, 192, 48, 192,
                                                     38L * 192, 48L * 192);
    pack_a_kernel<<<dim3(24, 4),  eblk, 0, stream>>>(dt_w, Ap_dt, 192, 6, 192, 32,
                                                     192L * 6, 192L * 32);
    pack_a_kernel<<<dim3(72, 1),  eblk, 0, stream>>>(out_w, Ap_ow, 96, 192, 96, 192, 0, 0);
    // ---- pack x into B panels ----
    pack_b_kernel<<<dim3(1536, 1), eblk, 0, stream>>>(x, Bp_x, 96, LPIX, 96, 0, 0);

    // 1) xz = in_conv_w(384x96) @ x(96x4096) + b
    wmma_gemm_f16_kernel<96, true><<<dim3(32, 24, 1), gblk, 0, stream>>>(
        Ap_in, Bp_x, xz, in_b, LPIX, 0, 0, 0);
    // 2) depthwise 3x3 + SiLU -> xg
    dwconv_silu_kernel<<<dim3(egrid), eblk, 0, stream>>>(xz, conv_w, conv_b, xg);
    // 3) build 4-direction xs (f32 rows + packed f16 panels)
    build_xs_kernel<<<dim3(egrid), eblk, 0, stream>>>(xg, xs, Bp_xs);
    // 4) channel means of xg
    chan_mean_kernel<<<dim3(24), eblk, 0, stream>>>(xg, cmean);
    // 5) x_dbl[k] (48 padded rows) = x_proj_w[k](48x192) @ xs[k](192x4096)
    wmma_gemm_f16_kernel<192, false><<<dim3(32, 3, 4), gblk, 0, stream>>>(
        Ap_xp, Bp_xs, xdbl, nullptr, LPIX, 48L * 192, 6L * LPIX * 32, 48L * LPIX);
    // 6) pack dt-rank rows (0..5) of x_dbl into zero-padded B panels
    pack_b_kernel<<<dim3(512, 4), eblk, 0, stream>>>(xdbl, Bp_dt, 6, LPIX, 32,
                                                     48L * LPIX, (long)LPIX * 32);
    // 7) dts[k] = dt_w[k](192x32pad) @ dtB[k](32x4096)
    wmma_gemm_f16_kernel<32, false><<<dim3(32, 12, 4), gblk, 0, stream>>>(
        Ap_dt, Bp_dt, dts, nullptr, LPIX, 192L * 32, (long)LPIX * 32, DL);
    // 8) selective scans (768 recurrences)
    scan_kernel<<<dim3(48), eblk, 0, stream>>>(xs, dts, xdbl, A_logs, Ds, dt_b, oy);
    // 9) combine directions
    combine_kernel<<<dim3(egrid), eblk, 0, stream>>>(oy, ycomb);
    // 10-12) channel branch
    chan_prep_kernel<<<dim3(1), eblk, 0, stream>>>(cmean, cin_w, cin_b, xcw, dtcw,
                                                   xcs, xcdbl, dtc);
    chan_scan_kernel<<<dim3(1), dim3(128), 0, stream>>>(xcs, xcdbl, dtc, Ac_logs, Dsc,
                                                        dtcb, yc);
    gate_kernel<<<dim3(1), eblk, 0, stream>>>(yc, cout_w, cout_b, cnw, cnb, gate);
    // 13) per-pixel LN * gate * silu(z) -> packed f16 panels
    ln_mult_kernel<<<dim3(512), eblk, 0, stream>>>(ycomb, onw, onb, gate, xz, Bp_yf);
    // 14) out = out_conv_w(96x192) @ yfin(192x4096) + b
    wmma_gemm_f16_kernel<192, true><<<dim3(32, 6, 1), gblk, 0, stream>>>(
        Ap_ow, Bp_yf, out, out_b, LPIX, 0, 0, 0);
}
